// SRRtransformer_68521908240909
// MI455X (gfx1250) — compile-verified, matrix operand order checked by reading, not run on hardware
//
#include <hip/hip_runtime.h>
#include <hip/hip_bf16.h>

// ---------------------------------------------------------------------------
// SRR transformer pipeline for MI455X (gfx1250, wave32, WMMA).
//  * One-time weight pre-pass: all GEMM/conv weights converted to f16 in
//    (N x Kpad) row-major layout (K zero-padded to a multiple of 32).
//  * Conv/GEMM kernel: v_wmma_f32_16x16x32_f16, 2 tiles/wave/K-step,
//    ds_load_b128 fragments, B-tile staged via global_load_async_to_lds_b128
//    (ASYNCcnt) when the builtin exists, plain b128 copy otherwise.
//  * Gather/scatter slice model + CG are L2-resident VALU kernels.
// Requires ~264 MB of d_ws scratch.
// ---------------------------------------------------------------------------

typedef __attribute__((ext_vector_type(16))) _Float16 v16h;
typedef __attribute__((ext_vector_type(8)))  _Float16 v8h;
typedef __attribute__((ext_vector_type(8)))  float    v8f;
typedef __attribute__((ext_vector_type(4)))  int      v4i;

#if defined(__has_builtin)
#  if __has_builtin(__builtin_amdgcn_global_load_async_to_lds_b128) && \
      __has_builtin(__builtin_amdgcn_s_wait_asynccnt)
#    define USE_ASYNC_LDS 1
#  endif
#endif
#ifndef USE_ASYNC_LDS
#  define USE_ASYNC_LDS 0
#endif

#if USE_ASYNC_LDS
typedef __attribute__((address_space(1))) v4i as1_v4i;
typedef __attribute__((address_space(3))) v4i as3_v4i;
#endif

#define NSL    64
#define HH     128
#define WW     128
#define DDZ    128
#define NPIX   (HH*WW)          // 16384
#define SLTOT  (NSL*NPIX)       // 1048576
#define VOLN   (DDZ*HH*WW)      // 2097152
#define DMODEL 256
#define DINNER 512
#define NLAYERS 4

// ============================ small helpers ================================

__device__ inline float trilerp(const float* __restrict__ vol, float x, float y, float z) {
    float x0f = floorf(x), y0f = floorf(y), z0f = floorf(z);
    float fx = x - x0f, fy = y - y0f, fz = z - z0f;
    int xi = (int)x0f, yi = (int)y0f, zi = (int)z0f;
    float acc = 0.f;
#pragma unroll
    for (int dz = 0; dz < 2; ++dz)
#pragma unroll
        for (int dy = 0; dy < 2; ++dy)
#pragma unroll
            for (int dx = 0; dx < 2; ++dx) {
                int X = xi + dx, Y = yi + dy, Z = zi + dz;
                if (X < 0 || X >= WW || Y < 0 || Y >= HH || Z < 0 || Z >= DDZ) continue;
                float w = (dx ? fx : 1.f - fx) * (dy ? fy : 1.f - fy) * (dz ? fz : 1.f - fz);
                acc += w * vol[(Z * HH + Y) * WW + X];
            }
    return acc;
}

__device__ inline void triscatter(float* __restrict__ vol, float x, float y, float z, float val) {
    float x0f = floorf(x), y0f = floorf(y), z0f = floorf(z);
    float fx = x - x0f, fy = y - y0f, fz = z - z0f;
    int xi = (int)x0f, yi = (int)y0f, zi = (int)z0f;
#pragma unroll
    for (int dz = 0; dz < 2; ++dz)
#pragma unroll
        for (int dy = 0; dy < 2; ++dy)
#pragma unroll
            for (int dx = 0; dx < 2; ++dx) {
                int X = xi + dx, Y = yi + dy, Z = zi + dz;
                if (X < 0 || X >= WW || Y < 0 || Y >= HH || Z < 0 || Z >= DDZ) continue;
                float w = (dx ? fx : 1.f - fx) * (dy ? fy : 1.f - fy) * (dz ? fz : 1.f - fz);
                atomicAdd(&vol[(Z * HH + Y) * WW + X], w * val);
            }
}

// ======================= mask = (gauss_blur(slices) > 0) ===================

__global__ void mask_kernel(const float* __restrict__ slices, float* __restrict__ mask) {
    int i = blockIdx.x * blockDim.x + threadIdx.x;
    if (i >= SLTOT) return;
    int n = i >> 14, rem = i & (NPIX - 1);
    int y = rem >> 7, x = rem & 127;
    const float w0 = 0.60653066f;
    float acc = 0.f;
    for (int dy = -1; dy <= 1; ++dy)
        for (int dx = -1; dx <= 1; ++dx) {
            int yy = y + dy, xx = x + dx;
            if (yy < 0 || yy >= HH || xx < 0 || xx >= WW) continue;
            float w = (dy ? w0 : 1.f) * (dx ? w0 : 1.f);
            acc += w * slices[(n << 14) + (yy << 7) + xx];
        }
    mask[i] = acc > 0.f ? 1.f : 0.f;
}

// ===================== slice acquisition (forward gather) ==================

__global__ void slice_acq_kernel(const float* __restrict__ T, const float* __restrict__ vol,
                                 const float* __restrict__ mask, const float* __restrict__ psf,
                                 const float* __restrict__ scale, float* __restrict__ out) {
    int i = blockIdx.x * blockDim.x + threadIdx.x;
    if (i >= SLTOT) return;
    int n = i >> 14, rem = i & (NPIX - 1);
    int y = rem >> 7, x = rem & 127;
    float m = mask ? mask[i] : 1.f;
    if (m == 0.f) { out[i] = 0.f; return; }
    float px = (float)x - 63.5f, py = (float)y - 63.5f;
    const float* Tr = T + n * 12;
    float r00 = Tr[0], r01 = Tr[1], r02 = Tr[2],  t0 = Tr[3];
    float r10 = Tr[4], r11 = Tr[5], r12 = Tr[6],  t1 = Tr[7];
    float r20 = Tr[8], r21 = Tr[9], r22 = Tr[10], t2 = Tr[11];
    float acc = 0.f;
    for (int kz = 0; kz < 3; ++kz)
        for (int ky = 0; ky < 3; ++ky)
            for (int kx = 0; kx < 3; ++kx) {
                float w = psf[(kz * 3 + ky) * 3 + kx];
                float sx = px + (float)(kx - 1);
                float sy = py + (float)(ky - 1);
                float sz = (float)(kz - 1);
                float wx = r00 * sx + r01 * sy + r02 * sz + t0 + 63.5f;
                float wy = r10 * sx + r11 * sy + r12 * sz + t1 + 63.5f;
                float wz = r20 * sx + r21 * sy + r22 * sz + t2 + 63.5f;
                acc += w * trilerp(vol, wx, wy, wz);
            }
    float s = scale ? scale[n] : 1.f;
    out[i] = acc * m * s;
}

// ==================== slice acquisition adjoint (scatter) ==================

__global__ void slice_adj_kernel(const float* __restrict__ T, const float* __restrict__ sl,
                                 const float* __restrict__ mask, const float* __restrict__ psf,
                                 const float* __restrict__ scale, float* __restrict__ volout) {
    int i = blockIdx.x * blockDim.x + threadIdx.x;
    if (i >= SLTOT) return;
    int n = i >> 14, rem = i & (NPIX - 1);
    int y = rem >> 7, x = rem & 127;
    float v = sl[i];
    if (mask) v *= mask[i];
    if (scale) v *= scale[n];
    if (v == 0.f) return;
    float px = (float)x - 63.5f, py = (float)y - 63.5f;
    const float* Tr = T + n * 12;
    float r00 = Tr[0], r01 = Tr[1], r02 = Tr[2],  t0 = Tr[3];
    float r10 = Tr[4], r11 = Tr[5], r12 = Tr[6],  t1 = Tr[7];
    float r20 = Tr[8], r21 = Tr[9], r22 = Tr[10], t2 = Tr[11];
    for (int kz = 0; kz < 3; ++kz)
        for (int ky = 0; ky < 3; ++ky)
            for (int kx = 0; kx < 3; ++kx) {
                float w = psf[(kz * 3 + ky) * 3 + kx];
                float sx = px + (float)(kx - 1);
                float sy = py + (float)(ky - 1);
                float sz = (float)(kz - 1);
                float wx = r00 * sx + r01 * sy + r02 * sz + t0 + 63.5f;
                float wy = r10 * sx + r11 * sy + r12 * sz + t1 + 63.5f;
                float wz = r20 * sx + r21 * sy + r22 * sz + t2 + 63.5f;
                triscatter(volout, wx, wy, wz, w * v);
            }
}

// ============================ channel concat ===============================

__global__ void concat_kernel(const float* __restrict__ a, const float* __restrict__ b,
                              float* __restrict__ z) {
    int i = blockIdx.x * blockDim.x + threadIdx.x;
    if (i >= SLTOT) return;
    int n = i >> 14, rem = i & (NPIX - 1);
    z[((n * 2 + 0) << 14) + rem] = a[i];
    z[((n * 2 + 1) << 14) + rem] = b[i];
}

// ==================== weight pre-pass: f32 -> f16 arena ====================

__global__ void cvt_pad_kernel(const float* __restrict__ src, _Float16* __restrict__ dst,
                               int N, int K, int Kpad) {
    int i = blockIdx.x * blockDim.x + threadIdx.x;
    if (i >= N * Kpad) return;
    int n = i / Kpad, k = i - n * Kpad;
    dst[i] = (k < K) ? (_Float16)src[(long)n * K + k] : (_Float16)0.f;
}

__global__ void cvtT_kernel(const float* __restrict__ src, _Float16* __restrict__ dst,
                            int K, int N) {
    int i = blockIdx.x * blockDim.x + threadIdx.x;
    if (i >= N * K) return;
    int n = i / K, k = i - n * K;
    dst[i] = (_Float16)src[(long)k * N + n];
}

// ================= generic implicit-GEMM conv via WMMA =====================

struct ConvParams {
    const float*    in;
    const _Float16* w;       // f16 arena, (Cout x Kpad) row-major
    const float*    bias;
    const float*    res;
    float*          out;
    int Nb, Cin, Hin, Win;
    int Cout;
    int Kpad;
    int spShift;    // log2(Hout*Wout)
    int woShift;    // log2(Wout)
    int stride, pad;
    int relu;
};

#define TM 64
#define TN 32
#define TK 32

template <int KS>
__global__ void __launch_bounds__(128) conv_gemm_wmma(ConvParams p) {
    const int sp   = 1 << p.spShift;
    const int Mtot = p.Nb << p.spShift;
    const int Ktot = p.Cin * KS * KS;
    const int m0 = blockIdx.y * TM;
    const int n0 = blockIdx.x * TN;
    const int tid = threadIdx.x;
    const int lane = tid & 31;
    const int wv = tid >> 5;
    const int woMask = (1 << p.woShift) - 1;

    __shared__ __align__(16) _Float16 As[TM][TK];     // 4 KB
    __shared__ __align__(16) _Float16 Bs[TN][TK];     // 2 KB, (n, k)

    v8f acc0 = {};
    v8f acc1 = {};

    const int mrow = wv * 16 + (lane & 15);
    const int ncol = lane & 15;
    const int khi  = (lane >> 4) << 3;      // 0 or 8

    // B-tile assignment: 4 threads per n-row, 16 bytes each
    const int bn  = tid >> 2;               // 0..31
    const int bsg = (tid & 3) << 3;         // 0,8,16,24 (halves)
    const _Float16* wrow = p.w + (long)(n0 + bn) * p.Kpad + bsg;

    for (int kk = 0; kk < p.Kpad; kk += TK) {
        // ---- cooperative im2col A-tile load (64x32 f16) ----
#pragma unroll
        for (int it = 0; it < (TM * TK) / 128; ++it) {
            int idx = it * 128 + tid;
            int mr = idx >> 5, kc = idx & 31;
            int m = m0 + mr, k = kk + kc;
            float v = 0.f;
            if (m < Mtot && k < Ktot) {
                int nb = m >> p.spShift;
                int rm = m & (sp - 1);
                int oy = rm >> p.woShift;
                int ox = rm & woMask;
                int ci = k / (KS * KS);
                int kr = k - ci * (KS * KS);
                int ky = kr / KS;
                int kx = kr - ky * KS;
                int iy = oy * p.stride + ky - p.pad;
                int ix = ox * p.stride + kx - p.pad;
                if (iy >= 0 && iy < p.Hin && ix >= 0 && ix < p.Win)
                    v = p.in[(((long)nb * p.Cin + ci) * p.Hin + iy) * p.Win + ix];
            }
            As[mr][kc] = (_Float16)v;
        }
        // ---- B-tile: one aligned 16B copy per thread (no bounds needed) ----
        {
            const _Float16* gsrc = wrow + kk;
#if USE_ASYNC_LDS
            __builtin_amdgcn_global_load_async_to_lds_b128(
                (as1_v4i*)(void*)gsrc,
                (as3_v4i*)(void*)&Bs[bn][bsg],
                0, 0);
#else
            *(v8h*)&Bs[bn][bsg] = *(const v8h*)gsrc;
#endif
            if (kk + TK < p.Kpad) __builtin_prefetch(gsrc + TK, 0, 3);
        }
#if USE_ASYNC_LDS
        __builtin_amdgcn_s_wait_asynccnt(0);
#endif
        __syncthreads();

        // ---- fragments: two ds_load_b128 each ----
        v8h a_lo = *(const v8h*)&As[mrow][khi];
        v8h a_hi = *(const v8h*)&As[mrow][16 + khi];
        v16h afrag = __builtin_shufflevector(a_lo, a_hi,
            0, 1, 2, 3, 4, 5, 6, 7, 8, 9, 10, 11, 12, 13, 14, 15);

        v8h b0_lo = *(const v8h*)&Bs[ncol][khi];
        v8h b0_hi = *(const v8h*)&Bs[ncol][16 + khi];
        v16h bfrag0 = __builtin_shufflevector(b0_lo, b0_hi,
            0, 1, 2, 3, 4, 5, 6, 7, 8, 9, 10, 11, 12, 13, 14, 15);

        v8h b1_lo = *(const v8h*)&Bs[16 + ncol][khi];
        v8h b1_hi = *(const v8h*)&Bs[16 + ncol][16 + khi];
        v16h bfrag1 = __builtin_shufflevector(b1_lo, b1_hi,
            0, 1, 2, 3, 4, 5, 6, 7, 8, 9, 10, 11, 12, 13, 14, 15);

        acc0 = __builtin_amdgcn_wmma_f32_16x16x32_f16(
            false, afrag, false, bfrag0, (short)0, acc0, false, false);
        acc1 = __builtin_amdgcn_wmma_f32_16x16x32_f16(
            false, afrag, false, bfrag1, (short)0, acc1, false, false);
        __syncthreads();
    }

    // ---- epilogue ----
#pragma unroll
    for (int v = 0; v < 8; ++v) {
        int m = m0 + wv * 16 + v + ((lane >> 4) << 3);
        if (m >= Mtot) continue;
        int nb = m >> p.spShift;
        int rm = m & (sp - 1);
#pragma unroll
        for (int t = 0; t < 2; ++t) {
            int n = n0 + t * 16 + (lane & 15);
            if (n >= p.Cout) continue;
            long oidx = (((long)nb * p.Cout + n) << p.spShift) + rm;
            float val = t ? acc1[v] : acc0[v];
            if (p.bias) val += p.bias[n];
            if (p.res) val += p.res[oidx];
            if (p.relu) val = fmaxf(val, 0.f);
            p.out[oidx] = val;
        }
    }
}

// ================= mean-pool (16x16) + positional embedding ================

__global__ void pool_pe_kernel(const float* __restrict__ hc, const float* __restrict__ theta,
                               const float* __restrict__ idx, const float* __restrict__ pos_w,
                               const float* __restrict__ pos_b, float* __restrict__ hs) {
    int n = blockIdx.x;
    int c = threadIdx.x;
    const float* base = hc + ((long)(n * DMODEL + c)) * 256;
    float s = 0.f;
    for (int i = 0; i < 256; ++i) s += base[i];
    s *= (1.f / 256.f);
    float pe = pos_b[c];
    for (int j = 0; j < 6; ++j) pe += theta[n * 6 + j] * pos_w[j * DMODEL + c];
    for (int j = 0; j < 2; ++j) pe += idx[n * 2 + j] * pos_w[(6 + j) * DMODEL + c];
    hs[n * DMODEL + c] = s + pe;
}

// ========================= attention (4 heads, 64 q) =======================

__global__ void attn_kernel(const float* __restrict__ q, const float* __restrict__ k,
                            const float* __restrict__ v, float* __restrict__ out) {
    int h = blockIdx.x;
    int qi = threadIdx.x;
    float s[NSL];
    float mx = -1e30f;
    for (int kj = 0; kj < NSL; ++kj) {
        float d = 0.f;
        for (int t = 0; t < 64; ++t)
            d += q[qi * DMODEL + h * 64 + t] * k[kj * DMODEL + h * 64 + t];
        d *= 0.125f;
        s[kj] = d;
        mx = fmaxf(mx, d);
    }
    float se = 0.f;
    for (int kj = 0; kj < NSL; ++kj) { s[kj] = expf(s[kj] - mx); se += s[kj]; }
    float inv = 1.f / se;
    for (int t = 0; t < 64; ++t) {
        float o = 0.f;
        for (int kj = 0; kj < NSL; ++kj) o += s[kj] * v[kj * DMODEL + h * 64 + t];
        out[qi * DMODEL + h * 64 + t] = o * inv;
    }
}

// ======================= fused residual-add + layernorm ====================

__global__ void add_ln_kernel(const float* __restrict__ x, const float* __restrict__ y,
                              const float* __restrict__ g, const float* __restrict__ b,
                              float* __restrict__ out) {
    __shared__ float red[DMODEL];
    int row = blockIdx.x, c = threadIdx.x;
    float v = x[row * DMODEL + c] + y[row * DMODEL + c];
    red[c] = v; __syncthreads();
    for (int s = DMODEL / 2; s > 0; s >>= 1) { if (c < s) red[c] += red[c + s]; __syncthreads(); }
    float mean = red[0] * (1.f / DMODEL); __syncthreads();
    float d = v - mean;
    red[c] = d * d; __syncthreads();
    for (int s = DMODEL / 2; s > 0; s >>= 1) { if (c < s) red[c] += red[c + s]; __syncthreads(); }
    float var = red[0] * (1.f / DMODEL);
    out[row * DMODEL + c] = d * rsqrtf(var + 1e-5f) * g[c] + b[c];
}

// ================= fc head + softmax(axis=0)*64, clamp 3 ===================

__global__ void head_kernel(const float* __restrict__ hs, const float* __restrict__ fc_w,
                            const float* __restrict__ fc_b, float* __restrict__ pvec,
                            float* __restrict__ xout) {
    __shared__ float sh[NSL];
    __shared__ float sh2[NSL];
    int n = threadIdx.x;
    float d = fc_b[0];
    for (int t = 0; t < DMODEL; ++t) d += hs[n * DMODEL + t] * fc_w[t];
    sh[n] = d; __syncthreads();
    float mx = -1e30f;
    for (int i = 0; i < NSL; ++i) mx = fmaxf(mx, sh[i]);
    float e = expf(d - mx);
    sh2[n] = e; __syncthreads();
    float se = 0.f;
    for (int i = 0; i < NSL; ++i) se += sh2[i];
    float xv = fminf(e / se * (float)NSL, 3.f);
    pvec[n] = xv;
    xout[n] = xv;
}

// ============================ CG primitives ================================

__global__ void fill_kernel(float* __restrict__ p, float v, int n) {
    int i = blockIdx.x * blockDim.x + threadIdx.x;
    if (i < n) p[i] = v;
}
__global__ void sub_kernel(const float* __restrict__ a, const float* __restrict__ b,
                           float* __restrict__ o, int n) {
    int i = blockIdx.x * blockDim.x + threadIdx.x;
    if (i < n) o[i] = a[i] - b[i];
}
__global__ void dot_kernel(const float* __restrict__ a, const float* __restrict__ b,
                           float* __restrict__ res, int n) {
    __shared__ float sh[256];
    int t = threadIdx.x;
    float v = 0.f;
    for (long j = blockIdx.x * 256 + t; j < n; j += (long)gridDim.x * 256) v += a[j] * b[j];
    sh[t] = v; __syncthreads();
    for (int s = 128; s > 0; s >>= 1) { if (t < s) sh[t] += sh[t + s]; __syncthreads(); }
    if (t == 0) atomicAdd(res, sh[0]);
}
__global__ void cg_xr_kernel(float* __restrict__ x, float* __restrict__ r,
                             const float* __restrict__ p, const float* __restrict__ Ap,
                             const float* __restrict__ rr, const float* __restrict__ pAp, int n) {
    int i = blockIdx.x * blockDim.x + threadIdx.x;
    if (i >= n) return;
    float alpha = rr[0] / pAp[0];
    x[i] += alpha * p[i];
    r[i] -= alpha * Ap[i];
}
__global__ void cg_p_kernel(float* __restrict__ p, const float* __restrict__ r,
                            const float* __restrict__ rr2, const float* __restrict__ rr, int n) {
    int i = blockIdx.x * blockDim.x + threadIdx.x;
    if (i >= n) return;
    float beta = rr2[0] / rr[0];
    p[i] = r[i] + beta * p[i];
}
__global__ void set_scalar_kernel(float* __restrict__ d, const float* __restrict__ s) {
    d[0] = s[0];
}
__global__ void relu_out_kernel(const float* __restrict__ x, float* __restrict__ o, int n) {
    int i = blockIdx.x * blockDim.x + threadIdx.x;
    if (i < n) o[i] = fmaxf(x[i], 0.f);
}

// =============================== host side =================================

static inline int ilog2(int v) { int s = 0; while ((1 << s) < v) ++s; return s; }

static inline void launch_conv(const float* in, const _Float16* w, const float* res, float* out,
                               int Nb, int Cin, int Hin, int Win, int Cout, int Hout, int Wout,
                               int ks, int stride, int pad, int relu, int Kpad, hipStream_t s) {
    ConvParams p{};
    p.in = in; p.w = w; p.bias = nullptr; p.res = res; p.out = out;
    p.Nb = Nb; p.Cin = Cin; p.Hin = Hin; p.Win = Win; p.Cout = Cout; p.Kpad = Kpad;
    p.spShift = ilog2(Hout * Wout); p.woShift = ilog2(Wout);
    p.stride = stride; p.pad = pad; p.relu = relu;
    int M = Nb * Hout * Wout;
    dim3 g((Cout + TN - 1) / TN, (M + TM - 1) / TM);
    if (ks == 7)      conv_gemm_wmma<7><<<g, 128, 0, s>>>(p);
    else if (ks == 3) conv_gemm_wmma<3><<<g, 128, 0, s>>>(p);
    else              conv_gemm_wmma<1><<<g, 128, 0, s>>>(p);
}

static inline void launch_gemm(const float* A, const _Float16* Bt, const float* bias, float* C,
                               int M, int N, int K, int relu, hipStream_t s) {
    ConvParams p{};
    p.in = A; p.w = Bt; p.bias = bias; p.res = nullptr; p.out = C;
    p.Nb = M; p.Cin = K; p.Hin = 1; p.Win = 1; p.Cout = N; p.Kpad = K;
    p.spShift = 0; p.woShift = 0; p.stride = 1; p.pad = 0; p.relu = relu;
    dim3 g((N + TN - 1) / TN, (M + TM - 1) / TM);
    conv_gemm_wmma<1><<<g, 128, 0, s>>>(p);
}

extern "C" void kernel_launch(void* const* d_in, const int* in_sizes, int n_in,
                              void* d_out, int out_size, void* d_ws, size_t ws_size,
                              hipStream_t stream) {
    (void)in_sizes; (void)n_in; (void)out_size; (void)ws_size;

    const float* theta   = (const float*)d_in[0];
    const float* T       = (const float*)d_in[1];
    const float* slices  = (const float*)d_in[2];
    const float* volume  = (const float*)d_in[3];
    const float* psf     = (const float*)d_in[4];
    const float* idx     = (const float*)d_in[5];
    const float* conv1_w = (const float*)d_in[6];
    const float* b1_w1   = (const float*)d_in[7];
    const float* b1_w2   = (const float*)d_in[8];
    const float* b2_w1   = (const float*)d_in[9];
    const float* b2_w2   = (const float*)d_in[10];
    const float* b2_proj = (const float*)d_in[11];
    const float* b3_w1   = (const float*)d_in[12];
    const float* b3_w2   = (const float*)d_in[13];
    const float* b3_proj = (const float*)d_in[14];
    const float* pos_w   = (const float*)d_in[15];
    const float* pos_b   = (const float*)d_in[16];
    const float* wq      = (const float*)d_in[17];
    const float* wk      = (const float*)d_in[18];
    const float* wv      = (const float*)d_in[19];
    const float* wo      = (const float*)d_in[20];
    const float* ln1_g   = (const float*)d_in[21];
    const float* ln1_b   = (const float*)d_in[22];
    const float* ff1_w   = (const float*)d_in[23];
    const float* ff1_b   = (const float*)d_in[24];
    const float* ff2_w   = (const float*)d_in[25];
    const float* ff2_b   = (const float*)d_in[26];
    const float* ln2_g   = (const float*)d_in[27];
    const float* ln2_b   = (const float*)d_in[28];
    const float* fc_w    = (const float*)d_in[29];
    const float* fc_b    = (const float*)d_in[30];

    float* out_vol = (float*)d_out;
    float* out_x   = out_vol + VOLN;

    const size_t MB = 1ull << 20;
    char* w8 = (char*)d_ws;
    float* mask  = (float*)(w8);                         // 4 MB
    float* est   = (float*)(w8 + 4 * MB);                // 4 MB
    float* hs    = (float*)(w8 + 8 * MB);                // small region
    float* qb    = (float*)(w8 + 8 * MB + 64 * 1024);
    float* kb    = (float*)(w8 + 8 * MB + 128 * 1024);
    float* vb    = (float*)(w8 + 8 * MB + 192 * 1024);
    float* ab    = (float*)(w8 + 8 * MB + 256 * 1024);
    float* pb    = (float*)(w8 + 8 * MB + 320 * 1024);
    float* fb    = (float*)(w8 + 8 * MB + 384 * 1024);
    float* pvec  = (float*)(w8 + 8 * MB + 768 * 1024);
    float* scal  = (float*)(w8 + 8 * MB + 800 * 1024);
    float* zcat  = (float*)(w8 + 10 * MB);               // 8 MB
    float* bufA  = (float*)(w8 + 18 * MB);               // 68 MB
    float* bufB  = (float*)(w8 + 86 * MB);               // 68 MB
    float* bufC  = (float*)(w8 + 154 * MB);              // 34 MB
    float* bufD  = (float*)(w8 + 188 * MB);              // 34 MB
    float* bufE  = (float*)(w8 + 222 * MB);              // 17 MB
    float* bufF  = (float*)(w8 + 239 * MB);              // 17 MB
    _Float16* wf = (_Float16*)(w8 + 256 * MB);           // 8 MB f16 weight arena

    // ---- f16 weight arena layout ----
    _Float16* c1f   = wf;                    // 64 x 128    (K=98 -> Kpad=128)
    _Float16* b1w1f = c1f   + 64  * 128;     // 64 x 576
    _Float16* b1w2f = b1w1f + 64  * 576;     // 64 x 576
    _Float16* b2w1f = b1w2f + 64  * 576;     // 128 x 576
    _Float16* b2w2f = b2w1f + 128 * 576;     // 128 x 1152
    _Float16* b2pf  = b2w2f + 128 * 1152;    // 128 x 64
    _Float16* b3w1f = b2pf  + 128 * 64;      // 256 x 1152
    _Float16* b3w2f = b3w1f + 256 * 1152;    // 256 x 2304
    _Float16* b3pf  = b3w2f + 256 * 2304;    // 256 x 128
    _Float16* wqT   = b3pf  + 256 * 128;     // 4 x (256 x 256)
    _Float16* wkT   = wqT   + NLAYERS * DMODEL * DMODEL;
    _Float16* wvT   = wkT   + NLAYERS * DMODEL * DMODEL;
    _Float16* woT   = wvT   + NLAYERS * DMODEL * DMODEL;
    _Float16* ff1T  = woT   + NLAYERS * DMODEL * DMODEL; // 4 x (512 x 256)
    _Float16* ff2T  = ff1T  + NLAYERS * DINNER * DMODEL; // 4 x (256 x 512)

    const int EB = 256;
    const int slblk = (SLTOT + EB - 1) / EB;
    const int volblk = (VOLN + EB - 1) / EB;
    auto cvt = [&](const float* s, _Float16* d, int N, int K, int Kpad) {
        int tot = N * Kpad;
        cvt_pad_kernel<<<(tot + EB - 1) / EB, EB, 0, stream>>>(s, d, N, K, Kpad);
    };
    auto cvtT = [&](const float* s, _Float16* d, int K, int N) {
        int tot = N * K;
        cvtT_kernel<<<(tot + EB - 1) / EB, EB, 0, stream>>>(s, d, K, N);
    };

    // ---- stage 0: weight pre-pass --------------------------------------
    cvt(conv1_w, c1f,   64, 98, 128);
    cvt(b1_w1,  b1w1f,  64, 576, 576);
    cvt(b1_w2,  b1w2f,  64, 576, 576);
    cvt(b2_w1,  b2w1f, 128, 576, 576);
    cvt(b2_w2,  b2w2f, 128, 1152, 1152);
    cvt(b2_proj, b2pf, 128, 64, 64);
    cvt(b3_w1,  b3w1f, 256, 1152, 1152);
    cvt(b3_w2,  b3w2f, 256, 2304, 2304);
    cvt(b3_proj, b3pf, 256, 128, 128);
    for (int l = 0; l < NLAYERS; ++l) {
        cvtT(wq + (long)l * DMODEL * DMODEL, wqT + (long)l * DMODEL * DMODEL, DMODEL, DMODEL);
        cvtT(wk + (long)l * DMODEL * DMODEL, wkT + (long)l * DMODEL * DMODEL, DMODEL, DMODEL);
        cvtT(wv + (long)l * DMODEL * DMODEL, wvT + (long)l * DMODEL * DMODEL, DMODEL, DMODEL);
        cvtT(wo + (long)l * DMODEL * DMODEL, woT + (long)l * DMODEL * DMODEL, DMODEL, DMODEL);
        cvtT(ff1_w + (long)l * DMODEL * DINNER, ff1T + (long)l * DINNER * DMODEL, DMODEL, DINNER);
        cvtT(ff2_w + (long)l * DINNER * DMODEL, ff2T + (long)l * DMODEL * DINNER, DINNER, DMODEL);
    }

    // ---- stage 1: mask + forward slice acquisition ----------------------
    mask_kernel<<<slblk, EB, 0, stream>>>(slices, mask);
    slice_acq_kernel<<<slblk, EB, 0, stream>>>(T, volume, mask, psf, nullptr, est);
    concat_kernel<<<slblk, EB, 0, stream>>>(slices, est, zcat);

    // ---- stage 2: CNN encoder (WMMA implicit GEMM) ----------------------
    launch_conv(zcat, c1f, nullptr, bufA, NSL, 2, 128, 128, 64, 64, 64, 7, 2, 2, 1, 128, stream);
    launch_conv(bufA, b1w1f, nullptr, bufB, NSL, 64, 64, 64, 64, 64, 64, 3, 1, 1, 1, 576, stream);
    launch_conv(bufB, b1w2f, bufA,   bufA, NSL, 64, 64, 64, 64, 64, 64, 3, 1, 1, 1, 576, stream);
    launch_conv(bufA, b2w1f, nullptr, bufC, NSL, 64, 64, 64, 128, 32, 32, 3, 2, 0, 1, 576, stream);
    launch_conv(bufC, b2w2f, nullptr, bufD, NSL, 128, 32, 32, 128, 32, 32, 3, 1, 1, 0, 1152, stream);
    launch_conv(bufA, b2pf,  bufD,   bufC, NSL, 64, 64, 64, 128, 32, 32, 1, 2, 0, 1, 64, stream);
    launch_conv(bufC, b3w1f, nullptr, bufE, NSL, 128, 32, 32, 256, 16, 16, 3, 2, 0, 1, 1152, stream);
    launch_conv(bufE, b3w2f, nullptr, bufF, NSL, 256, 16, 16, 256, 16, 16, 3, 1, 1, 0, 2304, stream);
    launch_conv(bufC, b3pf,  bufF,   bufE, NSL, 128, 32, 32, 256, 16, 16, 1, 2, 0, 1, 128, stream);

    // ---- stage 3: pooled features + positional embedding ----------------
    pool_pe_kernel<<<NSL, DMODEL, 0, stream>>>(bufE, theta, idx, pos_w, pos_b, hs);

    // ---- stage 4: transformer (4 layers) --------------------------------
    for (int l = 0; l < NLAYERS; ++l) {
        launch_gemm(hs, wqT + (long)l * DMODEL * DMODEL, nullptr, qb, NSL, DMODEL, DMODEL, 0, stream);
        launch_gemm(hs, wkT + (long)l * DMODEL * DMODEL, nullptr, kb, NSL, DMODEL, DMODEL, 0, stream);
        launch_gemm(hs, wvT + (long)l * DMODEL * DMODEL, nullptr, vb, NSL, DMODEL, DMODEL, 0, stream);
        attn_kernel<<<4, NSL, 0, stream>>>(qb, kb, vb, ab);
        launch_gemm(ab, woT + (long)l * DMODEL * DMODEL, nullptr, pb, NSL, DMODEL, DMODEL, 0, stream);
        add_ln_kernel<<<NSL, DMODEL, 0, stream>>>(hs, pb, ln1_g + l * DMODEL, ln1_b + l * DMODEL, hs);
        launch_gemm(hs, ff1T + (long)l * DINNER * DMODEL, ff1_b + l * DINNER, fb,
                    NSL, DINNER, DMODEL, 1, stream);
        launch_gemm(fb, ff2T + (long)l * DMODEL * DINNER, ff2_b + l * DMODEL, pb,
                    NSL, DMODEL, DINNER, 0, stream);
        add_ln_kernel<<<NSL, DMODEL, 0, stream>>>(hs, pb, ln2_g + l * DMODEL, ln2_b + l * DMODEL, hs);
    }

    // ---- stage 5: per-slice weights p -----------------------------------
    head_kernel<<<1, NSL, 0, stream>>>(hs, fc_w, fc_b, pvec, out_x);

    // ---- stage 6: CG solve (AtA x = b), 2 iterations --------------------
    float* bvol  = bufA;
    float* xv    = bufA + 2 * VOLN;
    float* rv    = bufA + 4 * VOLN;
    float* pvv   = bufA + 6 * VOLN;
    float* Apv   = bufA + 8 * VOLN;
    float* tmpsl = bufA + 10 * VOLN;
    float* rr  = scal + 0;
    float* pAp = scal + 1;
    float* rr2 = scal + 2;

    fill_kernel<<<volblk, EB, 0, stream>>>(bvol, 0.f, VOLN);
    slice_adj_kernel<<<slblk, EB, 0, stream>>>(T, slices, mask, psf, pvec, bvol);

    (void)hipMemcpyAsync(xv, volume, (size_t)VOLN * sizeof(float), hipMemcpyDeviceToDevice, stream);

    slice_acq_kernel<<<slblk, EB, 0, stream>>>(T, xv, mask, psf, pvec, tmpsl);
    fill_kernel<<<volblk, EB, 0, stream>>>(Apv, 0.f, VOLN);
    slice_adj_kernel<<<slblk, EB, 0, stream>>>(T, tmpsl, mask, psf, nullptr, Apv);
    sub_kernel<<<volblk, EB, 0, stream>>>(bvol, Apv, rv, VOLN);
    (void)hipMemcpyAsync(pvv, rv, (size_t)VOLN * sizeof(float), hipMemcpyDeviceToDevice, stream);
    fill_kernel<<<1, 1, 0, stream>>>(rr, 0.f, 1);
    dot_kernel<<<1024, 256, 0, stream>>>(rv, rv, rr, VOLN);

    for (int it = 0; it < 2; ++it) {
        slice_acq_kernel<<<slblk, EB, 0, stream>>>(T, pvv, mask, psf, pvec, tmpsl);
        fill_kernel<<<volblk, EB, 0, stream>>>(Apv, 0.f, VOLN);
        slice_adj_kernel<<<slblk, EB, 0, stream>>>(T, tmpsl, mask, psf, nullptr, Apv);
        fill_kernel<<<1, 1, 0, stream>>>(pAp, 0.f, 1);
        dot_kernel<<<1024, 256, 0, stream>>>(pvv, Apv, pAp, VOLN);
        cg_xr_kernel<<<volblk, EB, 0, stream>>>(xv, rv, pvv, Apv, rr, pAp, VOLN);
        fill_kernel<<<1, 1, 0, stream>>>(rr2, 0.f, 1);
        dot_kernel<<<1024, 256, 0, stream>>>(rv, rv, rr2, VOLN);
        cg_p_kernel<<<volblk, EB, 0, stream>>>(pvv, rv, rr2, rr, VOLN);
        set_scalar_kernel<<<1, 1, 0, stream>>>(rr, rr2);
    }

    // ---- stage 7: output = relu(vol), x ---------------------------------
    relu_out_kernel<<<volblk, EB, 0, stream>>>(xv, out_vol, VOLN);
}
